// TopKGate_20255065767969
// MI455X (gfx1250) — matrix operand order repaired and checked by reading, not compiled, not run on hardware
//
#include <hip/hip_runtime.h>

typedef __attribute__((ext_vector_type(16))) __bf16   v16bf;
typedef __attribute__((ext_vector_type(8)))  float    v8f;
typedef __attribute__((ext_vector_type(8)))  unsigned v8u;
typedef __attribute__((ext_vector_type(4)))  unsigned v4u;
typedef int v4i_vs __attribute__((__vector_size__(16)));   // matches builtin's int __vector(4)

#define DMODEL 4096
#define NEXP   64
#define ROWS   64            // token rows per block (4 waves x 16)
#define KC     128           // K elements per LDS chunk
#define SPD    68            // x-tile row stride in dwords (64 bf16-pairs + 4 pad)
#define NT     4             // 64 experts / 16
#define KSTEPS (KC / 32)     // WMMA K-steps per chunk
#define WFDW   (KSTEPS * NT * 32 * 8)   // 4096 dwords = 16KB of W frags per chunk
#define NCHUNK (DMODEL / KC)            // 32

#if __has_builtin(__builtin_amdgcn_global_load_async_to_lds_b128) && \
    __has_builtin(__builtin_amdgcn_s_wait_asynccnt)
#define HAVE_ASYNC_LDS 1
#else
#define HAVE_ASYNC_LDS 0
#endif

// round-to-nearest-even f32 -> bf16, packed pair
static __device__ __forceinline__ unsigned pack_bf16(float a, float b) {
  unsigned ua = __float_as_uint(a);
  unsigned ub = __float_as_uint(b);
  unsigned ra = (ua + 0x7FFFu + ((ua >> 16) & 1u)) >> 16;
  unsigned rb = (ub + 0x7FFFu + ((ub >> 16) & 1u)) >> 16;
  return (rb << 16) | (ra & 0xFFFFu);
}

// ---- prep: build bf16 W image in B-fragment layout in workspace ----
// dword d within chunk c:  d = (((ks*NT+nt)*32 + lane)*8 + v),  global dword = c*WFDW + d
// fragment: lane holds N = nt*16+(lane&15); dword v holds K = {2v+16h, 2v+16h+1}
__global__ __launch_bounds__(128, 1) void prep_wfrag_kernel(
    const float* __restrict__ W, unsigned* __restrict__ wfg) {
  int pid = blockIdx.x * 128 + threadIdx.x;   // one uint2 (dword pair) each; 131072 launched (image needs 65536)
  int c   = pid >> 11;                        // 2048 pairs per chunk
  int e   = pid & 2047;
  int vp  = e & 3;
  int l   = (e >> 2) & 31;
  int nt  = (e >> 7) & 3;
  int ks  = e >> 9;
  int n   = nt * 16 + (l & 15);
  int K0  = c * KC + ks * 32 + vp * 4 + (l >> 4) * 16;
  const float4 wr = *(const float4*)(W + (size_t)n * DMODEL + K0);
  *(uint2*)(wfg + (size_t)pid * 2) = make_uint2(pack_bf16(wr.x, wr.y), pack_bf16(wr.z, wr.w));
}

template <bool USE_WS>
__global__ __launch_bounds__(128, 1) void topk_gate_kernel(
    const float* __restrict__ x, const float* __restrict__ W,
    const float* __restrict__ b, float* __restrict__ out,
    const unsigned* __restrict__ wfg) {
  __shared__ __attribute__((aligned(16))) unsigned XT[ROWS * SPD];  // x tile (bf16 pairs), reused for f32 scores
  __shared__ __attribute__((aligned(32))) unsigned WF[WFDW];        // W B-fragments

  const int t    = threadIdx.x;
  const int wv   = t >> 5;
  const int lane = t & 31;
  const int h    = lane >> 4;
  const int m16  = lane & 15;
  const int row0 = blockIdx.x * ROWS;

  v8f acc[NT];
#pragma unroll
  for (int i = 0; i < NT; ++i) acc[i] = (v8f){0.f, 0.f, 0.f, 0.f, 0.f, 0.f, 0.f, 0.f};

  for (int kb = 0; kb < DMODEL; kb += KC) {
    __syncthreads();  // previous chunk's LDS reads complete before overwrite

    // ---- stage W fragments ----
    if (USE_WS) {
      const unsigned* wfgc = wfg + (size_t)(kb / KC) * WFDW;
#pragma unroll
      for (int i = 0; i < 8; ++i) {
        int e = t + i * 128;  // 1024 x 16B
#if HAVE_ASYNC_LDS
        __builtin_amdgcn_global_load_async_to_lds_b128(
            (__attribute__((address_space(1))) v4i_vs*)(uintptr_t)(wfgc + e * 4),
            (__attribute__((address_space(3))) v4i_vs*)(unsigned)(uintptr_t)&WF[e * 4],
            0, 0);
#else
        *(uint4*)&WF[e * 4] = *(const uint4*)(wfgc + e * 4);
#endif
      }
    } else {
#pragma unroll
      for (int i = 0; i < 16; ++i) {
        int e  = t + i * 128;
        int vp = e & 3;
        int l  = (e >> 2) & 31;
        int nt = (e >> 7) & 3;
        int ks = e >> 9;
        int n  = nt * 16 + (l & 15);
        int K0 = kb + ks * 32 + vp * 4 + (l >> 4) * 16;
        const float4 wr = *(const float4*)(W + (size_t)n * DMODEL + K0);
        *(uint2*)&WF[(((ks * NT + nt) << 5) + l) * 8 + vp * 2] =
            make_uint2(pack_bf16(wr.x, wr.y), pack_bf16(wr.z, wr.w));
      }
    }

    // ---- stage x tile: coalesced b128 loads, cvt to bf16 pairs ----
#pragma unroll
    for (int i = 0; i < 16; ++i) {
      int j   = t + i * 128;  // 2048 float4s = 64 rows x 32
      int row = j >> 5;
      int c4  = j & 31;
      const float4 xv = *(const float4*)(x + (size_t)(row0 + row) * DMODEL + kb + c4 * 4);
      *(uint2*)&XT[row * SPD + c4 * 2] = make_uint2(pack_bf16(xv.x, xv.y), pack_bf16(xv.z, xv.w));
    }

    // ---- prefetch next x chunk (global_prefetch_b8) ----
    if (kb + KC < DMODEL) {
      const float* pf = x + (size_t)(row0 + (t >> 1)) * DMODEL + (kb + KC) + (t & 1) * 64;
      __builtin_prefetch(pf, 0, 0);
      __builtin_prefetch(pf + 32, 0, 0);
    }

#if HAVE_ASYNC_LDS
    if (USE_WS) __builtin_amdgcn_s_wait_asynccnt(0);
#endif
    __syncthreads();

    // ---- compute: 16 WMMAs per chunk per wave ----
    const unsigned* xt = &XT[(wv * 16 + m16) * SPD];
#pragma unroll
    for (int ks = 0; ks < KSTEPS; ++ks) {
      const v4u a0 = *(const v4u*)&xt[ks * 16 + h * 8];
      const v4u a1 = *(const v4u*)&xt[ks * 16 + h * 8 + 4];
      v8u au;
      au[0] = a0[0]; au[1] = a0[1]; au[2] = a0[2]; au[3] = a0[3];
      au[4] = a1[0]; au[5] = a1[1]; au[6] = a1[2]; au[7] = a1[3];
      const v16bf afrag = __builtin_bit_cast(v16bf, au);
#pragma unroll
      for (int nt = 0; nt < NT; ++nt) {
        const v8u bu = *(const v8u*)&WF[(((ks * NT + nt) << 5) + lane) * 8];
        const v16bf bfrag = __builtin_bit_cast(v16bf, bu);
        acc[nt] = __builtin_amdgcn_wmma_f32_16x16x32_bf16(
            false, afrag, false, bfrag, (short)0, acc[nt], false, false);
      }
    }
  }

  // ---- dump scores to LDS (each wave writes only its own 16 rows) ----
  float* SF = (float*)XT;
#pragma unroll
  for (int nt = 0; nt < NT; ++nt)
#pragma unroll
    for (int v = 0; v < 8; ++v) {
      int row = wv * 16 + v + h * 8;  // C/D layout: VGPR v -> M = v (+8 for upper half)
      SF[row * SPD + nt * 16 + m16] = acc[nt][v];
    }
  __syncthreads();

  // ---- gating epilogue: one thread per token row ----
  if (t < ROWS) {
    const float* s = &SF[t * SPD];
    float m1 = -INFINITY, m2 = -INFINITY;
    int i1 = 0, i2 = 0;
    for (int c = 0; c < NEXP; ++c) {
      float sc = s[c] + b[c];
      if (sc > m1) { m2 = m1; i2 = i1; m1 = sc; i1 = c; }
      else if (sc > m2) { m2 = sc; i2 = c; }
    }
    float Z = 0.f;
    for (int c = 0; c < NEXP; ++c) Z += __expf(s[c] + b[c] - m1);
    float e2 = __expf(m2 - m1);
    float w1 = 1.0f / Z;
    float w2 = e2 / Z;
    float ws = w1 + w2 + 1e-8f;
    float o1 = w1 / ws, o2 = w2 / ws;

    float4* orow = (float4*)(out + (size_t)(row0 + t) * NEXP);
#pragma unroll
    for (int c4 = 0; c4 < 16; ++c4) {
      const int c = c4 * 4;
      float4 o;
      o.x = (c + 0 == i1) ? o1 : (c + 0 == i2) ? o2 : 0.f;
      o.y = (c + 1 == i1) ? o1 : (c + 1 == i2) ? o2 : 0.f;
      o.z = (c + 2 == i1) ? o1 : (c + 2 == i2) ? o2 : 0.f;
      o.w = (c + 3 == i1) ? o1 : (c + 3 == i2) ? o2 : 0.f;
      orow[c4] = o;
    }
  }
}

extern "C" void kernel_launch(void* const* d_in, const int* in_sizes, int n_in,
                              void* d_out, int out_size, void* d_ws, size_t ws_size,
                              hipStream_t stream) {
  const float* x = (const float*)d_in[0];
  const float* W = (const float*)d_in[1];
  const float* b = (const float*)d_in[2];
  float* out = (float*)d_out;
  const int T = in_sizes[0] / DMODEL;                 // 16384
  const size_t wf_bytes = (size_t)NCHUNK * WFDW * 4;  // 512 KB

  if (ws_size >= wf_bytes) {
    unsigned* wfg = (unsigned*)d_ws;
    prep_wfrag_kernel<<<(NEXP * DMODEL / 2) / 128, 128, 0, stream>>>(W, wfg);
    topk_gate_kernel<true><<<T / ROWS, 128, 0, stream>>>(x, W, b, out, wfg);
  } else {
    topk_gate_kernel<false><<<T / ROWS, 128, 0, stream>>>(x, W, b, out, nullptr);
  }
}
